// BiencoderRanker_48722109006492
// MI455X (gfx1250) — matrix operand order repaired and checked by reading, not compiled, not run on hardware
//
#include <hip/hip_runtime.h>
#include <stdint.h>

typedef __attribute__((ext_vector_type(16))) _Float16 v16h;
typedef __attribute__((ext_vector_type(8)))  _Float16 v8h;
typedef __attribute__((ext_vector_type(4)))  _Float16 v4h;
typedef __attribute__((ext_vector_type(8)))  float    v8f;
typedef __attribute__((ext_vector_type(4)))  float    v4f;

constexpr int KDIM = 4096;          // reduction length (fixed by reference)
constexpr int BN   = 128;           // pred rows per workgroup
constexpr int BM   = 256;           // cand rows per workgroup
constexpr int BK   = 32;            // K chunk (= one WMMA K depth)
constexpr int KSTEPS = KDIM / BK;   // 128

// ---- f16-in-LDS fast path tiling ----
constexpr int HSTRIDE = 40;                       // padded LDS row stride in halves (80B: 20 banks, no conflicts, 16B aligned)
constexpr int A_BUF_HALFS = BN * HSTRIDE;         // 5120
constexpr int B_BUF_HALFS = BM * HSTRIDE;         // 10240
constexpr size_t LDS_H_BYTES = (size_t)(2 * A_BUF_HALFS + 2 * B_BUF_HALFS) * 2;   // 61440 B

// ---- f32-in-LDS fallback tiling (round-1 design) ----
constexpr int FSTRIDE = 36;                       // padded stride in floats (144B)
constexpr int A_BUF_FLOATS = BN * FSTRIDE;        // 4608
constexpr int B_BUF_FLOATS = BM * FSTRIDE;        // 9216
constexpr size_t LDS_F_BYTES = (size_t)(2 * A_BUF_FLOATS + 2 * B_BUF_FLOATS) * 4; // 110592 B

// gfx1250 async global->LDS copy (ASYNCcnt path). Per-lane 16B; dest VGPR holds the
// workgroup-relative LDS byte offset (low 32 bits of the flat shared address).
__device__ __forceinline__ void async_copy_b128(uint32_t lds_off, const void* gsrc) {
  asm volatile("global_load_async_to_lds_b128 %0, %1, off"
               :: "v"(lds_off), "v"(gsrc) : "memory");
}

// ---------------- fused row inv-norm + raw f16 pack: one wave32 per row ----------------
__global__ __launch_bounds__(256)
void pack_rows_kernel(const float* __restrict__ X, _Float16* __restrict__ Xh,
                      float* __restrict__ inv_norm, int rows) {
  const int lane = threadIdx.x & 31;
  const int wave = threadIdx.x >> 5;
  const int row  = blockIdx.x * 8 + wave;
  if (row >= rows) return;
  const float4* p = (const float4*)(X + (size_t)row * KDIM);
  v4h* ph = (v4h*)(Xh + (size_t)row * KDIM);
  float s = 0.f;
  for (int i = lane; i < KDIM / 4; i += 32) {
    float4 v = p[i];
    s += v.x * v.x + v.y * v.y + v.z * v.z + v.w * v.w;
    v4h h;
    h[0] = (_Float16)v.x; h[1] = (_Float16)v.y; h[2] = (_Float16)v.z; h[3] = (_Float16)v.w;
    ph[i] = h;
  }
  #pragma unroll
  for (int off = 16; off > 0; off >>= 1) s += __shfl_xor(s, off, 32);
  if (lane == 0) inv_norm[row] = 1.0f / fmaxf(sqrtf(s), 1e-8f);  // matches torch eps clamp
}

// ---------------- norms only (fallback path) ----------------
__global__ __launch_bounds__(256)
void rownorm_kernel(const float* __restrict__ X, float* __restrict__ inv_norm, int rows) {
  const int lane = threadIdx.x & 31;
  const int wave = threadIdx.x >> 5;
  const int row  = blockIdx.x * 8 + wave;
  if (row >= rows) return;
  const float4* p = (const float4*)(X + (size_t)row * KDIM);
  float s = 0.f;
  for (int i = lane; i < KDIM / 4; i += 32) {
    float4 v = p[i];
    s += v.x * v.x + v.y * v.y + v.z * v.z + v.w * v.w;
  }
  #pragma unroll
  for (int off = 16; off > 0; off >>= 1) s += __shfl_xor(s, off, 32);
  if (lane == 0) inv_norm[row] = 1.0f / fmaxf(sqrtf(s), 1e-8f);
}

// ================= FAST PATH: f16 tiles, zero conversions in the K loop =================
__global__ __launch_bounds__(256, 2)
void cosine_gemm_f16_kernel(const _Float16* __restrict__ Ah,   // (N, K) pred, raw f16
                            const _Float16* __restrict__ Bh,   // (M, K) cand, raw f16
                            const float* __restrict__ invA,
                            const float* __restrict__ invB,
                            float* __restrict__ out, int Mtot) {
  extern __shared__ _Float16 ldsH[];
  _Float16* sA = ldsH;                      // [2][BN][HSTRIDE]
  _Float16* sB = ldsH + 2 * A_BUF_HALFS;    // [2][BM][HSTRIDE]

  const int t    = threadIdx.x;
  const int lane = t & 31;
  const int wv   = t >> 5;
  const int wrow = wv >> 2;     // 0..1
  const int wcol = wv & 3;      // 0..3
  const int l16  = lane & 15;
  const int lhi  = lane >> 4;

  const int n0 = blockIdx.x * BN;   // x fastest => 8 N-tiles sharing a B stripe adjacent (L2 reuse)
  const int m0 = blockIdx.y * BM;

  const uint32_t sA_base = (uint32_t)(uintptr_t)sA;
  const uint32_t sB_base = (uint32_t)(uintptr_t)sB;

  // 6 async b128 issues per thread per K-step: A 128x32 f16 (2/thread) + B 256x32 f16 (4/thread)
  auto issue = [&](int buf, int kc) {
    #pragma unroll
    for (int i = 0; i < 2; ++i) {
      const int idx = t + 256 * i;            // 0..511
      const int row = idx >> 2, c = idx & 3;  // c = which 8-half (16B) chunk
      const _Float16* g = Ah + (size_t)(n0 + row) * KDIM + kc + c * 8;
      const uint32_t l = sA_base + (uint32_t)((buf * A_BUF_HALFS + row * HSTRIDE + c * 8) * 2);
      async_copy_b128(l, g);
    }
    #pragma unroll
    for (int i = 0; i < 4; ++i) {
      const int idx = t + 256 * i;            // 0..1023
      const int row = idx >> 2, c = idx & 3;
      const _Float16* g = Bh + (size_t)(m0 + row) * KDIM + kc + c * 8;
      const uint32_t l = sB_base + (uint32_t)((buf * B_BUF_HALFS + row * HSTRIDE + c * 8) * 2);
      async_copy_b128(l, g);
    }
  };

  v8f acc[4][4] = {};

  issue(0, 0);
  int kc = 0;
  for (int s = 0; s < KSTEPS; ++s, kc += BK) {
    const int cur = s & 1;
    if (s + 1 < KSTEPS) {
      issue(cur ^ 1, kc + BK);
      asm volatile("s_wait_asynccnt 6" ::: "memory");   // previous 6 (current buffer) done
    } else {
      asm volatile("s_wait_asynccnt 0" ::: "memory");
    }
    __syncthreads();

    const _Float16* bA = sA + cur * A_BUF_HALFS;
    const _Float16* bB = sB + cur * B_BUF_HALFS;

    v16h af[4], bf[4];
    // A frag (16-bit A 16x32): lane L: M=L&15; lanes0-15 hold K 0-7 & 16-23, lanes16-31 K 8-15 & 24-31
    #pragma unroll
    for (int i = 0; i < 4; ++i) {
      const v8h* p = (const v8h*)(bA + (wrow * 64 + i * 16 + l16) * HSTRIDE + lhi * 8);
      const v8h x = p[0];   // K kb..kb+7
      const v8h y = p[2];   // K kb+16..kb+23
      af[i] = __builtin_shufflevector(x, y, 0,1,2,3,4,5,6,7, 8,9,10,11,12,13,14,15);
    }
    // B frag (16-bit B 32x16): lane L: N=L&15; lanes0-15 hold K 0-7 & 16-23, lanes16-31 K 8-15 & 24-31
    #pragma unroll
    for (int j = 0; j < 4; ++j) {
      const v8h* p = (const v8h*)(bB + (wcol * 64 + j * 16 + l16) * HSTRIDE + lhi * 8);
      const v8h x = p[0];
      const v8h y = p[2];
      bf[j] = __builtin_shufflevector(x, y, 0,1,2,3,4,5,6,7, 8,9,10,11,12,13,14,15);
    }

    #pragma unroll
    for (int i = 0; i < 4; ++i)
      #pragma unroll
      for (int j = 0; j < 4; ++j)
        acc[i][j] = __builtin_amdgcn_wmma_f32_16x16x32_f16(
            false, af[i], false, bf[j], (short)0, acc[i][j], false, false);

    __syncthreads();
  }

  #pragma unroll
  for (int i = 0; i < 4; ++i) {
    const int gn = n0 + wrow * 64 + i * 16 + lhi * 8;
    float ia[8];
    #pragma unroll
    for (int r = 0; r < 8; ++r) ia[r] = invA[gn + r];
    #pragma unroll
    for (int j = 0; j < 4; ++j) {
      const int gm = m0 + wcol * 64 + j * 16 + l16;
      const float ib = invB[gm];
      float* o = out + (size_t)gn * Mtot + gm;
      #pragma unroll
      for (int r = 0; r < 8; ++r)
        o[(size_t)r * Mtot] = acc[i][j][r] * ia[r] * ib;
    }
  }
}

// ================= FALLBACK: f32 tiles, convert at fragment time (round-1 design) =================
__global__ __launch_bounds__(256, 2)
void cosine_gemm_f32lds_kernel(const float* __restrict__ A, const float* __restrict__ B,
                               const float* __restrict__ invA, const float* __restrict__ invB,
                               float* __restrict__ out, int Mtot) {
  extern __shared__ float lds[];
  float* sA = lds;
  float* sB = lds + 2 * A_BUF_FLOATS;

  const int t    = threadIdx.x;
  const int lane = t & 31;
  const int wv   = t >> 5;
  const int wrow = wv >> 2;
  const int wcol = wv & 3;
  const int l16  = lane & 15;
  const int lhi  = lane >> 4;

  const int n0 = blockIdx.x * BN;
  const int m0 = blockIdx.y * BM;

  const uint32_t sA_base = (uint32_t)(uintptr_t)sA;
  const uint32_t sB_base = (uint32_t)(uintptr_t)sB;

  auto issue = [&](int buf, int kc) {
    #pragma unroll
    for (int i = 0; i < 4; ++i) {
      const int idx = t + 256 * i;
      const int row = idx >> 3, c4 = idx & 7;
      const float* g = A + (size_t)(n0 + row) * KDIM + kc + c4 * 4;
      const uint32_t l = sA_base + (uint32_t)((buf * A_BUF_FLOATS + row * FSTRIDE + c4 * 4) * 4);
      async_copy_b128(l, g);
    }
    #pragma unroll
    for (int i = 0; i < 8; ++i) {
      const int idx = t + 256 * i;
      const int row = idx >> 3, c4 = idx & 7;
      const float* g = B + (size_t)(m0 + row) * KDIM + kc + c4 * 4;
      const uint32_t l = sB_base + (uint32_t)((buf * B_BUF_FLOATS + row * FSTRIDE + c4 * 4) * 4);
      async_copy_b128(l, g);
    }
  };

  v8f acc[4][4] = {};

  issue(0, 0);
  int kc = 0;
  for (int s = 0; s < KSTEPS; ++s, kc += BK) {
    const int cur = s & 1;
    if (s + 1 < KSTEPS) {
      issue(cur ^ 1, kc + BK);
      asm volatile("s_wait_asynccnt 12" ::: "memory");
    } else {
      asm volatile("s_wait_asynccnt 0" ::: "memory");
    }
    __syncthreads();

    const float* bA = sA + cur * A_BUF_FLOATS;
    const float* bB = sB + cur * B_BUF_FLOATS;

    v16h af[4], bf[4];
    #pragma unroll
    for (int i = 0; i < 4; ++i) {
      const v4f* p = (const v4f*)(bA + (wrow * 64 + i * 16 + l16) * FSTRIDE + lhi * 8);
      const v4f q0 = p[0], q1 = p[1], q2 = p[4], q3 = p[5];
      v16h h;
      #pragma unroll
      for (int e = 0; e < 4; ++e) {
        h[e] = (_Float16)q0[e]; h[4 + e] = (_Float16)q1[e];
        h[8 + e] = (_Float16)q2[e]; h[12 + e] = (_Float16)q3[e];
      }
      af[i] = h;
    }
    #pragma unroll
    for (int j = 0; j < 4; ++j) {
      const v4f* p = (const v4f*)(bB + (wcol * 64 + j * 16 + l16) * FSTRIDE + lhi * 8);
      const v4f q0 = p[0], q1 = p[1], q2 = p[4], q3 = p[5];
      v16h h;
      #pragma unroll
      for (int e = 0; e < 4; ++e) {
        h[e] = (_Float16)q0[e]; h[4 + e] = (_Float16)q1[e];
        h[8 + e] = (_Float16)q2[e]; h[12 + e] = (_Float16)q3[e];
      }
      bf[j] = h;
    }

    #pragma unroll
    for (int i = 0; i < 4; ++i)
      #pragma unroll
      for (int j = 0; j < 4; ++j)
        acc[i][j] = __builtin_amdgcn_wmma_f32_16x16x32_f16(
            false, af[i], false, bf[j], (short)0, acc[i][j], false, false);

    __syncthreads();
  }

  #pragma unroll
  for (int i = 0; i < 4; ++i) {
    const int gn = n0 + wrow * 64 + i * 16 + lhi * 8;
    float ia[8];
    #pragma unroll
    for (int r = 0; r < 8; ++r) ia[r] = invA[gn + r];
    #pragma unroll
    for (int j = 0; j < 4; ++j) {
      const int gm = m0 + wcol * 64 + j * 16 + l16;
      const float ib = invB[gm];
      float* o = out + (size_t)gn * Mtot + gm;
      #pragma unroll
      for (int r = 0; r < 8; ++r)
        o[(size_t)r * Mtot] = acc[i][j][r] * ia[r] * ib;
    }
  }
}

extern "C" void kernel_launch(void* const* d_in, const int* in_sizes, int n_in,
                              void* d_out, int out_size, void* d_ws, size_t ws_size,
                              hipStream_t stream) {
  const float* pred = (const float*)d_in[0];   // (N, 4096) f32
  const float* cand = (const float*)d_in[1];   // (M, 4096) f32
  float* out = (float*)d_out;                  // (N, M) f32

  const int N = in_sizes[0] / KDIM;            // 1024
  const int M = in_sizes[1] / KDIM;            // 16384

  float* invP = (float*)d_ws;                  // N floats
  float* invC = invP + N;                      // M floats

  const size_t normBytes = (size_t)(N + M) * sizeof(float);
  const size_t halfOff   = (normBytes + 255) & ~(size_t)255;           // 16B+ aligned
  const size_t halfBytes = (size_t)(N + M) * (size_t)KDIM * 2;         // ~143 MB
  const bool fast = (ws_size >= halfOff + halfBytes);                  // deterministic per run

  dim3 grid(N / BN, M / BM);                   // (8, 64); shapes divide tiles exactly

  if (fast) {
    _Float16* predH = (_Float16*)((char*)d_ws + halfOff);
    _Float16* candH = predH + (size_t)N * KDIM;
    pack_rows_kernel<<<(N + 7) / 8, 256, 0, stream>>>(pred, predH, invP, N);
    pack_rows_kernel<<<(M + 7) / 8, 256, 0, stream>>>(cand, candH, invC, M);
    cosine_gemm_f16_kernel<<<grid, 256, LDS_H_BYTES, stream>>>(predH, candH, invP, invC, out, M);
  } else {
    rownorm_kernel<<<(N + 7) / 8, 256, 0, stream>>>(pred, invP, N);
    rownorm_kernel<<<(M + 7) / 8, 256, 0, stream>>>(cand, invC, M);
    cosine_gemm_f32lds_kernel<<<grid, 256, LDS_F_BYTES, stream>>>(pred, cand, invP, invC, out, M);
  }
}
